// QLSTM_65481071403955
// MI455X (gfx1250) — compile-verified, matrix operand order checked by reading, not run on hardware
//
#include <hip/hip_runtime.h>

// ---------------------------------------------------------------------------
// QLSTM for MI455X (gfx1250).
// Key observation: gates are (B,1) scalars broadcast over H and cx0==0, so
// the whole hidden state collapses to one scalar per batch element. The only
// real FLOPs are X @ W[:, :D].T  (M=131072, K=128, N=32), done with
// v_wmma_f32_16x16x32_f16 using a hi/lo f16 split for ~fp32 accuracy.
// ---------------------------------------------------------------------------

#define BQ   128           // batch
#define TQ   1024          // seq len
#define DQ   128           // input dim
#define HQ   256           // hidden dim
#define NQ   32            // 4 gates * 8 qubits
#define MQ   (BQ * TQ)     // GEMM rows

typedef __attribute__((ext_vector_type(16))) _Float16 v16h;
typedef __attribute__((ext_vector_type(8)))  float    v8f;

// Workspace layout in floats:
//   [0,      4096) : Wpack  (32 x 128)   packed gate weights, input part
//   [4096,   4128) : WhSum  (32)         rowsum of W[:, D:D+H]
//   [4128,   4160) : bias   (32)
//   [4160,   4192) : qw     (32)
//   [4192,   4224) : rz     (32)
//   [4224,   4256) : cnot   (32)         padded, q==7 -> 0
//   [8192,   8192 + M*32)      : A  (GEMM result)
//   [.. +M)                    : hbuf (hval per (b,t))
//   [.. +128)                  : cbuf (final cval per b)
#define WS_WHS   4096
#define WS_BIAS  4128
#define WS_QW    4160
#define WS_RZ    4192
#define WS_CNOT  4224
#define WS_A     8192

// ---------------------------------------------------------------------------
// Pack weights/params into workspace.
// ---------------------------------------------------------------------------
__global__ void qlstm_pack(const float* Wf, const float* bf, const float* qwf,
                           const float* rzf, const float* cf,
                           const float* Wi, const float* bi, const float* qwi,
                           const float* rzi, const float* ci,
                           const float* Wg, const float* bg, const float* qwg,
                           const float* rzg, const float* cg,
                           const float* Wo, const float* bo, const float* qwo,
                           const float* rzo, const float* co,
                           float* __restrict__ P) {
  const float* W[4]  = {Wf, Wi, Wg, Wo};
  const float* bb[4] = {bf, bi, bg, bo};
  const float* qw[4] = {qwf, qwi, qwg, qwo};
  const float* rz[4] = {rzf, rzi, rzg, rzo};
  const float* cn[4] = {cf, ci, cg, co};

  int t = threadIdx.x;
  // Wpack[n][k] = W_gate[q][k]  for k < D
  for (int idx = t; idx < NQ * DQ; idx += blockDim.x) {
    int n = idx >> 7, k = idx & 127;
    int g = n >> 3, q = n & 7;
    P[idx] = W[g][q * (DQ + HQ) + k];
  }
  if (t < NQ) {
    int g = t >> 3, q = t & 7;
    float s = 0.f;
    for (int j = 0; j < HQ; ++j) s += W[g][q * (DQ + HQ) + DQ + j];
    P[WS_WHS  + t] = s;
    P[WS_BIAS + t] = bb[g][q];
    P[WS_QW   + t] = qw[g][q];
    P[WS_RZ   + t] = rz[g][q];
    P[WS_CNOT + t] = (q < 7) ? cn[g][q] : 0.f;
  }
}

// Load 16 floats (k..k+7 and k+16..k+23) and split into f16 hi/lo halves.
__device__ __forceinline__ void load_split(const float* __restrict__ p,
                                           v16h& hi, v16h& lo) {
  float buf[16];
  float4 v0 = *(const float4*)(p);
  float4 v1 = *(const float4*)(p + 4);
  float4 v2 = *(const float4*)(p + 16);
  float4 v3 = *(const float4*)(p + 20);
  buf[0]=v0.x; buf[1]=v0.y; buf[2]=v0.z; buf[3]=v0.w;
  buf[4]=v1.x; buf[5]=v1.y; buf[6]=v1.z; buf[7]=v1.w;
  buf[8]=v2.x; buf[9]=v2.y; buf[10]=v2.z; buf[11]=v2.w;
  buf[12]=v3.x; buf[13]=v3.y; buf[14]=v3.z; buf[15]=v3.w;
#pragma unroll
  for (int i = 0; i < 16; ++i) {
    _Float16 h = (_Float16)buf[i];
    hi[i] = h;
    lo[i] = (_Float16)(buf[i] - (float)h);
  }
}

// ---------------------------------------------------------------------------
// GEMM: A[m][n] = sum_k x[m][k] * Wpack[n][k], M=131072, K=128, N=32.
// One wave per 16-row M tile, both 16-col N tiles. 24 WMMAs per wave.
// ---------------------------------------------------------------------------
__global__ __launch_bounds__(128)
void qlstm_gemm(const float* __restrict__ x, const float* __restrict__ P,
                float* __restrict__ A) {
  int wave  = threadIdx.x >> 5;
  int lane  = threadIdx.x & 31;
  int mtile = blockIdx.x * 4 + wave;
  int m0    = mtile * 16;
  int r     = lane & 15;
  int kg    = (lane >> 4) * 8;      // K sub-offset for this half-wave

  const float* xrow = x + (size_t)(m0 + r) * DQ;
  const float* b0r  = P + (size_t)r * DQ;          // W rows 0..15
  const float* b1r  = P + (size_t)(16 + r) * DQ;   // W rows 16..31

  v8f acc0 = {};
  v8f acc1 = {};

#pragma unroll
  for (int c = 0; c < 4; ++c) {
    int kb = 32 * c + kg;
    v16h ahi, alo, bhi, blo;
    load_split(xrow + kb, ahi, alo);

    load_split(b0r + kb, bhi, blo);
    acc0 = __builtin_amdgcn_wmma_f32_16x16x32_f16(false, ahi, false, bhi,
                                                  (short)0, acc0, false, false);
    acc0 = __builtin_amdgcn_wmma_f32_16x16x32_f16(false, ahi, false, blo,
                                                  (short)0, acc0, false, false);
    acc0 = __builtin_amdgcn_wmma_f32_16x16x32_f16(false, alo, false, bhi,
                                                  (short)0, acc0, false, false);

    load_split(b1r + kb, bhi, blo);
    acc1 = __builtin_amdgcn_wmma_f32_16x16x32_f16(false, ahi, false, bhi,
                                                  (short)0, acc1, false, false);
    acc1 = __builtin_amdgcn_wmma_f32_16x16x32_f16(false, ahi, false, blo,
                                                  (short)0, acc1, false, false);
    acc1 = __builtin_amdgcn_wmma_f32_16x16x32_f16(false, alo, false, bhi,
                                                  (short)0, acc1, false, false);
  }

  // C/D layout: VGPR i -> row i (lanes 0-15) / row i+8 (lanes 16-31), col = lane&15
  int rowadd = (lane >> 4) * 8;
  int col    = lane & 15;
#pragma unroll
  for (int i = 0; i < 8; ++i) {
    size_t m = (size_t)(m0 + rowadd + i);
    A[m * NQ + col]      = acc0[i];
    A[m * NQ + 16 + col] = acc1[i];
  }
}

// ---------------------------------------------------------------------------
// Sequential scan: one wave per batch element, one lane per (gate,qubit).
// ---------------------------------------------------------------------------
__device__ __forceinline__ float fast_sigmoid(float x) {
  return 1.f / (1.f + __expf(-x));
}
__device__ __forceinline__ float fast_tanh(float x) {
  float e = __expf(2.f * x);
  return (e - 1.f) / (e + 1.f);
}

__global__ __launch_bounds__(32)
void qlstm_scan(const float* __restrict__ P, const float* __restrict__ A,
                float* __restrict__ hbuf, float* __restrict__ cbuf) {
  int b    = blockIdx.x;
  int lane = threadIdx.x;          // 0..31 : n = gate*8 + q
  int gsel = lane >> 3;

  float whs  = P[WS_WHS  + lane];
  float bias = P[WS_BIAS + lane];
  float qw   = P[WS_QW   + lane];
  float rz   = P[WS_RZ   + lane];
  float cn   = P[WS_CNOT + lane];  // 0 for q==7

  const float* Ab = A + (size_t)b * TQ * NQ;
  float hval = 0.f, cval = 0.f;

  // 8-deep register ring buffer keeps A loads off the critical chain.
  float abuf[8];
#pragma unroll
  for (int i = 0; i < 8; ++i) abuf[i] = Ab[(size_t)i * NQ + lane];

  for (int t = 0; t < TQ; ++t) {
    float a = abuf[t & 7];
    if (t + 8 < TQ) abuf[t & 7] = Ab[(size_t)(t + 8) * NQ + lane];

    float p = a + hval * whs + bias;
    // cos(z)cos(rz) - sin(z)sin(rz) == cos(z + rz)
    float z = __cosf(p * qw + rz);
    // CNOT ladder: z[q] += cnot[q]*z[q+1]  (cn==0 masks q==7 / gate boundary)
    float zn = __shfl_down(z, 1, 32);
    float zl = z + cn * zn;
    // mean over the 8 lanes of this gate
    zl += __shfl_xor(zl, 1, 32);
    zl += __shfl_xor(zl, 2, 32);
    zl += __shfl_xor(zl, 4, 32);
    float gv = zl * 0.125f;

    float sg  = fast_sigmoid(gv);
    float th  = fast_tanh(gv);
    float act = (gsel == 2) ? th : sg;

    float f  = __shfl(act, 0,  32);
    float ii = __shfl(act, 8,  32);
    float gg = __shfl(act, 16, 32);
    float oo = __shfl(act, 24, 32);

    cval = f * cval + ii * gg;
    hval = oo * fast_tanh(cval);
    if (lane == 0) hbuf[(size_t)b * TQ + t] = hval;
  }
  if (lane == 0) cbuf[b] = cval;
}

// ---------------------------------------------------------------------------
// Broadcast hval over H into ys, plus final hx / cx. float4 stores.
// ---------------------------------------------------------------------------
__global__ __launch_bounds__(256)
void qlstm_bcast(const float* __restrict__ hbuf, const float* __restrict__ cbuf,
                 float* __restrict__ out) {
  const size_t YS = (size_t)BQ * TQ * HQ;   // 33554432
  const size_t HX = (size_t)BQ * HQ;        // 32768
  size_t idx = ((size_t)blockIdx.x * blockDim.x + threadIdx.x) * 4;
  if (idx >= YS + 2 * HX) return;

  float v;
  if (idx < YS) {
    v = hbuf[idx >> 8];                          // (b*T+t) = idx / H
  } else if (idx < YS + HX) {
    size_t bb = (idx - YS) >> 8;
    v = hbuf[bb * TQ + (TQ - 1)];                // hx = last hval
  } else {
    size_t bb = (idx - YS - HX) >> 8;
    v = cbuf[bb];                                // cx
  }
  float4 o4 = make_float4(v, v, v, v);
  *(float4*)(out + idx) = o4;
}

// ---------------------------------------------------------------------------
extern "C" void kernel_launch(void* const* d_in, const int* in_sizes, int n_in,
                              void* d_out, int out_size, void* d_ws, size_t ws_size,
                              hipStream_t stream) {
  const float* x = (const float*)d_in[0];
  // per-gate params: W, b, qw, rz, cnot at 1 + 5*j
  const float* Wf  = (const float*)d_in[1];
  const float* bf  = (const float*)d_in[2];
  const float* qwf = (const float*)d_in[3];
  const float* rzf = (const float*)d_in[4];
  const float* cf  = (const float*)d_in[5];
  const float* Wi  = (const float*)d_in[6];
  const float* bi  = (const float*)d_in[7];
  const float* qwi = (const float*)d_in[8];
  const float* rzi = (const float*)d_in[9];
  const float* ci  = (const float*)d_in[10];
  const float* Wg  = (const float*)d_in[11];
  const float* bg  = (const float*)d_in[12];
  const float* qwg = (const float*)d_in[13];
  const float* rzg = (const float*)d_in[14];
  const float* cg  = (const float*)d_in[15];
  const float* Wo  = (const float*)d_in[16];
  const float* bo  = (const float*)d_in[17];
  const float* qwo = (const float*)d_in[18];
  const float* rzo = (const float*)d_in[19];
  const float* co  = (const float*)d_in[20];

  float* P    = (float*)d_ws;
  float* A    = P + WS_A;
  float* hbuf = A + (size_t)MQ * NQ;
  float* cbuf = hbuf + MQ;
  float* out  = (float*)d_out;

  qlstm_pack<<<1, 256, 0, stream>>>(Wf, bf, qwf, rzf, cf,
                                    Wi, bi, qwi, rzi, ci,
                                    Wg, bg, qwg, rzg, cg,
                                    Wo, bo, qwo, rzo, co, P);
  qlstm_gemm<<<MQ / 64, 128, 0, stream>>>(x, P, A);
  qlstm_scan<<<BQ, 32, 0, stream>>>(P, A, hbuf, cbuf);
  size_t total4 = ((size_t)BQ * TQ * HQ + 2 * (size_t)BQ * HQ) / 4;
  qlstm_bcast<<<(total4 + 255) / 256, 256, 0, stream>>>(hbuf, cbuf, out);
}
